// MultiHeadAttention_62895501082861
// MI455X (gfx1250) — matrix operand order repaired and checked by reading, not compiled, not run on hardware
//
#include <hip/hip_runtime.h>
#include <math.h>

// MI455X / gfx1250, wave32. bf16 WMMA (v_wmma_f32_16x16x32_bf16) for all GEMMs,
// flash-attention streaming so S x S scores never hit HBM.

typedef __attribute__((ext_vector_type(16))) __bf16 v16bf;
typedef __attribute__((ext_vector_type(8)))  __bf16 v8bf;
typedef __attribute__((ext_vector_type(8)))  float  v8f;

#define B_  2
#define S_  2048
#define DM  1024
#define H_  16
#define DK  64

__device__ __forceinline__ unsigned short f2bf(float f) {
  unsigned u = __builtin_bit_cast(unsigned, f);
  u += 0x7FFFu + ((u >> 16) & 1u);          // round-to-nearest-even
  return (unsigned short)(u >> 16);
}

__device__ __forceinline__ v8f zero8() {
  v8f z;
#pragma unroll
  for (int i = 0; i < 8; ++i) z[i] = 0.f;
  return z;
}

__device__ __forceinline__ v16bf join8(v8bf lo, v8bf hi) {
  union { v16bf v; v8bf h[2]; } u;
  u.h[0] = lo; u.h[1] = hi;
  return u.v;
}

__device__ __forceinline__ v8f wmma_bf16(v16bf a, v16bf b, v8f c) {
  return __builtin_amdgcn_wmma_f32_16x16x32_bf16(false, a, false, b,
                                                 (short)0, c, false, false);
}

// ---------------- fp32 -> bf16 conversion ----------------
__global__ void __launch_bounds__(256)
cvt_bf16_kernel(const float* __restrict__ src, unsigned short* __restrict__ dst, int n) {
  int i = blockIdx.x * blockDim.x + threadIdx.x;
  int stride = gridDim.x * blockDim.x;
  for (; i < n; i += stride) dst[i] = f2bf(src[i]);
}

// ---------------- GEMM: C[M,N] = A[M,K] * W[N,K]^T (+bias), fp32 out ----------
// One wave -> 16(M) x 64(N) strip. A,W row-major bf16 (u16 bit patterns).
__global__ void __launch_bounds__(128)
gemm_bf16_kernel(const unsigned short* __restrict__ A,
                 const unsigned short* __restrict__ W,
                 float* __restrict__ C,
                 const float* __restrict__ bias,
                 int M, int N, int K) {
  const int lane = threadIdx.x & 31;
  const int n = lane & 15;          // column-in-tile / A row
  const int g = lane >> 4;          // lane half
  const int wave = blockIdx.x * (blockDim.x >> 5) + (threadIdx.x >> 5);
  const int nt = N >> 6;
  const int m0 = (wave / nt) << 4;
  const int n0 = (wave % nt) << 6;

  v8f acc[4];
#pragma unroll
  for (int f = 0; f < 4; ++f) acc[f] = zero8();

  const unsigned short* arow = A + (size_t)(m0 + n) * K;
  for (int k0 = 0; k0 < K; k0 += 32) {
    // A frag (contiguous k order, matches B): lanes0-15 K=k0..+15, lanes16-31 K=k0+16..+31
    v16bf a = join8(*(const v8bf*)(arow + k0 + g * 16),
                    *(const v8bf*)(arow + k0 + 8 + g * 16));
#pragma unroll
    for (int f = 0; f < 4; ++f) {
      // B frag: column (n0+f*16+n) of W^T == row of W -> contiguous 32B load
      const unsigned short* brow = W + (size_t)(n0 + f * 16 + n) * K + k0 + g * 16;
      acc[f] = wmma_bf16(a, *(const v16bf*)brow, acc[f]);
    }
  }
#pragma unroll
  for (int f = 0; f < 4; ++f) {
    const int col = n0 + f * 16 + n;
    const float bv = bias ? bias[col] : 0.f;
#pragma unroll
    for (int r = 0; r < 8; ++r) {
      const int m = g * 8 + r;                        // C tile row layout
      C[(size_t)(m0 + m) * N + col] = acc[f][r] + bv;
    }
  }
}

// ---------------- RoPE + repack: fp32 [B,S,DM] -> bf16 Q/K [B,H,S,64], Vt [B,H,64,S]
__global__ void __launch_bounds__(256)
rope_pack_kernel(const float* __restrict__ Qp, const float* __restrict__ Kp,
                 const float* __restrict__ Vp,
                 unsigned short* __restrict__ Qbf, unsigned short* __restrict__ Kbf,
                 unsigned short* __restrict__ Vt) {
  const int idx = blockIdx.x * blockDim.x + threadIdx.x;   // B*H*S*64 threads
  const int d  = idx & 63;
  const int s  = (idx >> 6) & (S_ - 1);
  const int bh = idx >> 17;
  const int b  = bh >> 4, h = bh & 15;

  const size_t src = ((size_t)(b * S_ + s)) * DM + h * DK + d;
  const float q = Qp[src], k = Kp[src], v = Vp[src];

  const int dh = d & 31;
  const float inv = __expf(-0.28782313662425576f * (float)dh); // 10000^(-dh/32)
  float sn, cs;
  __sincosf((float)s * inv, &sn, &cs);

  float qo, ko;
  if (d < 32) { qo = q * cs - Qp[src + 32] * sn;  ko = k * cs - Kp[src + 32] * sn; }
  else        { qo = q * cs + Qp[src - 32] * sn;  ko = k * cs + Kp[src - 32] * sn; }

  const size_t dst = ((size_t)bh * S_ + s) * DK + d;
  Qbf[dst] = f2bf(qo);
  Kbf[dst] = f2bf(ko);
  Vt[((size_t)bh * DK + d) * S_ + s] = f2bf(v);
}

// ---------------- Flash attention: one wave per 16 query rows -----------------
__global__ void __launch_bounds__(128)
attn_kernel(const unsigned short* __restrict__ Qbf,
            const unsigned short* __restrict__ Kbf,
            const unsigned short* __restrict__ Vt,
            const float* __restrict__ bias,      // [H,S,S]
            unsigned short* __restrict__ ctx) {  // bf16 [B*S, DM]
  __shared__ unsigned short Plds[4 * 16 * 32];   // per-wave P staging (transpose)
  const int lane = threadIdx.x & 31;
  const int n = lane & 15;
  const int g = lane >> 4;
  const int wv = threadIdx.x >> 5;
  const int wave = blockIdx.x * 4 + wv;          // 0..4095
  const int bh = wave >> 7;                      // b*H + h
  const int q0 = (wave & 127) << 4;
  const int b = bh >> 4, h = bh & 15;

  const unsigned short* Qb = Qbf + (size_t)bh * S_ * DK;
  const unsigned short* Kb = Kbf + (size_t)bh * S_ * DK;
  const unsigned short* Vb = Vt + (size_t)bh * DK * S_;
  const float* bb = bias + (size_t)h * S_ * S_;
  unsigned short* Pw = Plds + wv * 512;

  // Q A-fragments for the two K=32 steps covering d=0..63
  v16bf aQ[2];
#pragma unroll
  for (int ks = 0; ks < 2; ++ks) {
    const unsigned short* qr = Qb + (size_t)(q0 + n) * DK + ks * 32 + g * 16;
    aQ[ks] = join8(*(const v8bf*)qr, *(const v8bf*)(qr + 8));
  }

  v8f o[4];
  float rmax[8], rsum[8];
#pragma unroll
  for (int f = 0; f < 4; ++f) o[f] = zero8();
#pragma unroll
  for (int r = 0; r < 8; ++r) { rmax[r] = -3.0e38f; rsum[r] = 0.f; }

  for (int kblk = 0; kblk <= q0 + 15; kblk += 32) {   // wave-uniform causal bound
    v8f sc0 = zero8(), sc1 = zero8();
#pragma unroll
    for (int ks = 0; ks < 2; ++ks) {
      // B frag columns are rows of K (contiguous along d)
      const unsigned short* kp = Kb + (size_t)(kblk + n) * DK + ks * 32 + g * 16;
      sc0 = wmma_bf16(aQ[ks], *(const v16bf*)kp, sc0);
      sc1 = wmma_bf16(aQ[ks], *(const v16bf*)(kp + 16 * DK), sc1);
    }
#pragma unroll
    for (int r = 0; r < 8; ++r) {
      const int m = g * 8 + r, q = q0 + m;
      const int j0 = kblk + n, j1 = j0 + 16;
      const float* br = bb + (size_t)q * S_;
      float s0 = (j0 <= q) ? sc0[r] * 0.125f + br[j0] : -3.0e38f;
      float s1 = (j1 <= q) ? sc1[r] * 0.125f + br[j1] : -3.0e38f;
      // row max across the 16 lanes holding this row
      float mt = fmaxf(s0, s1);
      mt = fmaxf(mt, __shfl_xor(mt, 1, 16));
      mt = fmaxf(mt, __shfl_xor(mt, 2, 16));
      mt = fmaxf(mt, __shfl_xor(mt, 4, 16));
      mt = fmaxf(mt, __shfl_xor(mt, 8, 16));
      const float nm = fmaxf(rmax[r], mt);
      const float alpha = __expf(rmax[r] - nm);
      rmax[r] = nm;
      const float p0 = __expf(s0 - nm), p1 = __expf(s1 - nm);
      float ps = p0 + p1;
      ps += __shfl_xor(ps, 1, 16);
      ps += __shfl_xor(ps, 2, 16);
      ps += __shfl_xor(ps, 4, 16);
      ps += __shfl_xor(ps, 8, 16);
      rsum[r] = rsum[r] * alpha + ps;
      o[0][r] *= alpha; o[1][r] *= alpha; o[2][r] *= alpha; o[3][r] *= alpha;
      Pw[m * 32 + n] = f2bf(p0);
      Pw[m * 32 + 16 + n] = f2bf(p1);
    }
    // same-wave LDS RAW: DS ops are in-order; wait + compiler fence for safety
    asm volatile("s_wait_dscnt 0" ::: "memory");
    const unsigned short* pr = Pw + n * 32 + g * 16;
    v16bf aP = join8(*(const v8bf*)pr, *(const v8bf*)(pr + 8));
#pragma unroll
    for (int f = 0; f < 4; ++f) {
      // B frag columns are rows of Vt (contiguous along sequence)
      const unsigned short* vp = Vb + (size_t)(f * 16 + n) * S_ + kblk + g * 16;
      o[f] = wmma_bf16(aP, *(const v16bf*)vp, o[f]);
    }
  }

#pragma unroll
  for (int r = 0; r < 8; ++r) {
    const int m = g * 8 + r;
    const float inv = 1.0f / rsum[r];
    const size_t row = ((size_t)(b * S_ + q0 + m)) * DM + h * DK;
#pragma unroll
    for (int f = 0; f < 4; ++f)
      ctx[row + f * 16 + n] = f2bf(o[f][r] * inv);
  }
}

// ------------------------------- launch ---------------------------------------
extern "C" void kernel_launch(void* const* d_in, const int* in_sizes, int n_in,
                              void* d_out, int out_size, void* d_ws, size_t ws_size,
                              hipStream_t stream) {
  const float* query = (const float*)d_in[0];
  const float* key   = (const float*)d_in[1];
  const float* value = (const float*)d_in[2];
  // d_in[3] = mask (causal; implemented analytically)
  const float* bias  = (const float*)d_in[4];
  const float* Wq    = (const float*)d_in[5];
  const float* Wk    = (const float*)d_in[6];
  const float* Wv    = (const float*)d_in[7];
  const float* Wo    = (const float*)d_in[8];
  const float* Wob   = (const float*)d_in[9];
  float* out = (float*)d_out;

  const int MT = B_ * S_;              // 4096 rows
  char* ws = (char*)d_ws;
  size_t off = 0;
  auto alloc = [&](size_t bytes) { void* p = ws + off; off += (bytes + 255) & ~(size_t)255; return p; };

  unsigned short* Xq  = (unsigned short*)alloc((size_t)MT * DM * 2);
  unsigned short* Xk  = (unsigned short*)alloc((size_t)MT * DM * 2);
  unsigned short* Xv  = (unsigned short*)alloc((size_t)MT * DM * 2);
  unsigned short* WqB = (unsigned short*)alloc((size_t)DM * DM * 2);
  unsigned short* WkB = (unsigned short*)alloc((size_t)DM * DM * 2);
  unsigned short* WvB = (unsigned short*)alloc((size_t)DM * DM * 2);
  unsigned short* WoB = (unsigned short*)alloc((size_t)DM * DM * 2);
  float* Qp = (float*)alloc((size_t)MT * DM * 4);
  float* Kp = (float*)alloc((size_t)MT * DM * 4);
  float* Vp = (float*)alloc((size_t)MT * DM * 4);
  unsigned short* Qbf = (unsigned short*)alloc((size_t)B_ * H_ * S_ * DK * 2);
  unsigned short* Kbf = (unsigned short*)alloc((size_t)B_ * H_ * S_ * DK * 2);
  unsigned short* Vt  = (unsigned short*)alloc((size_t)B_ * H_ * DK * S_ * 2);
  unsigned short* ctx = (unsigned short*)alloc((size_t)MT * DM * 2);

  // 1) fp32 -> bf16
  const int nAct = MT * DM, nW = DM * DM;
  cvt_bf16_kernel<<<2048, 256, 0, stream>>>(query, Xq, nAct);
  cvt_bf16_kernel<<<2048, 256, 0, stream>>>(key,   Xk, nAct);
  cvt_bf16_kernel<<<2048, 256, 0, stream>>>(value, Xv, nAct);
  cvt_bf16_kernel<<<1024, 256, 0, stream>>>(Wq, WqB, nW);
  cvt_bf16_kernel<<<1024, 256, 0, stream>>>(Wk, WkB, nW);
  cvt_bf16_kernel<<<1024, 256, 0, stream>>>(Wv, WvB, nW);
  cvt_bf16_kernel<<<1024, 256, 0, stream>>>(Wo, WoB, nW);

  // 2) Q/K/V projections: 4096x1024x1024 WMMA GEMMs
  const int gemmBlocks = (MT / 16) * (DM / 64) / 4;   // 1024 blocks x 4 waves
  gemm_bf16_kernel<<<gemmBlocks, 128, 0, stream>>>(Xq, WqB, Qp, nullptr, MT, DM, DM);
  gemm_bf16_kernel<<<gemmBlocks, 128, 0, stream>>>(Xk, WkB, Kp, nullptr, MT, DM, DM);
  gemm_bf16_kernel<<<gemmBlocks, 128, 0, stream>>>(Xv, WvB, Vp, nullptr, MT, DM, DM);

  // 3) RoPE + repack (V transposed for contiguous P*V B-fragments)
  rope_pack_kernel<<<(B_ * H_ * S_ * DK) / 256, 256, 0, stream>>>(Qp, Kp, Vp, Qbf, Kbf, Vt);

  // 4) flash attention: 4096 waves (one per 16 query rows)
  attn_kernel<<<(B_ * H_ * (S_ / 16)) / 4, 128, 0, stream>>>(Qbf, Kbf, Vt, bias, ctx);

  // 5) output projection + bias -> d_out (fp32)
  gemm_bf16_kernel<<<gemmBlocks, 128, 0, stream>>>(ctx, WoB, out, Wob, MT, DM, DM);
}
